// ProjectGraphMemory_6554120093978
// MI455X (gfx1250) — matrix operand order repaired and checked by reading, not compiled, not run on hardware
//
#include <hip/hip_runtime.h>
#include <hip/hip_bf16.h>
#include <stdint.h>

// GATv2 x2 + proj + mean-pool for MI455X (gfx1250, wave32).
// GEMMs: LDS-staged tiles -> v_wmma_f32_16x16x32_f16 (f16 operands, f32 accumulate).
// Edge phase = L2-resident gathers + fp32 atomics (node tables fit in 192MB L2).

typedef __attribute__((ext_vector_type(16))) _Float16 v16h;
typedef __attribute__((ext_vector_type(8)))  _Float16 v8h;
typedef __attribute__((ext_vector_type(8)))  float    v8f;

// ---------- helpers ----------
__device__ __forceinline__ unsigned f32key(float f) {
  unsigned b = __float_as_uint(f);
  return b ^ ((b & 0x80000000u) ? 0xFFFFFFFFu : 0x80000000u); // order-preserving
}
__device__ __forceinline__ float keyf32(unsigned k) {
  unsigned b = (k & 0x80000000u) ? (k ^ 0x80000000u) : ~k;
  return __uint_as_float(b);
}

// ---------- generic zero ----------
__global__ void zero_u32_kernel(unsigned* __restrict__ p, long long n) {
  long long i = (long long)blockIdx.x * blockDim.x + threadIdx.x;
  if (i < n) p[i] = 0u;
}

// ---------- self-loop attr (fill_value='mean') ----------
__global__ void loop_sum_kernel(const int* __restrict__ dst, const float* __restrict__ ea,
                                float* __restrict__ s, float* __restrict__ cnt, int E) {
  int e = blockIdx.x * blockDim.x + threadIdx.x;
  if (e >= E) return;
  int d = dst[e];
  atomicAdd(s + d, ea[e]);
  atomicAdd(cnt + d, 1.0f);
}
__global__ void loop_div_kernel(const float* __restrict__ s, const float* __restrict__ cnt,
                                float* __restrict__ loopa, int N) {
  int n = blockIdx.x * blockDim.x + threadIdx.x;
  if (n >= N) return;
  float c = cnt[n];
  loopa[n] = (c > 0.0f) ? s[n] / fmaxf(c, 1.0f) : 0.0f;
}

// ---------- WMMA GEMM: C[M,Nc] = A[M,K] @ W[K,Nc] + bias[Nc] ----------
// Requires K % 32 == 0, Nc % 32 == 0, M % 16 == 0 (true here: M=50000=3125*16,
// K in {128,256,64}, Nc in {256,64,32}).
// Block = 64 threads (2 waves); block tile = 16(M) x 32(N); K stepped by 32.
// Tiles staged in LDS (f16); B stored in WMMA fragment order so each lane's
// fragment is a contiguous 32B LDS read. A and B both use contiguous k = 16*half + elem.
__global__ __launch_bounds__(64)
void gemm_wmma_kernel(const float* __restrict__ A, const float* __restrict__ W,
                      const float* __restrict__ bias, float* __restrict__ C,
                      int M, int K, int Nc) {
  __shared__ _Float16 sA[16 * 32];        // [m][k] row-major (1 KB)
  __shared__ _Float16 sB[2 * 32 * 16];    // [wave][lane][elem] fragment order (2 KB)

  const int t    = threadIdx.x;
  const int wave = t >> 5;                 // 0..1  -> N sub-tile
  const int lane = t & 31;
  const int half = lane >> 4;              // 0: lanes 0-15, 1: lanes 16-31
  const int l16  = lane & 15;
  const int m0   = blockIdx.y * 16;
  const int n0   = blockIdx.x * 32;

  // A cooperative load: thread t loads 8 floats (two float4) of row (t>>2), cols (t&3)*8..+7
  const int ra = t >> 2;                   // 0..15
  const int ca = (t & 3) * 8;              // 0,8,16,24
  const int arow = (m0 + ra < M) ? (m0 + ra) : (M - 1);
  const float* Aptr = A + (size_t)arow * K + ca;

  v8f acc = {};

  for (int k0 = 0; k0 < K; k0 += 32) {
    // ---- global loads (coalesced b128) ----
    float4 a0 = *(const float4*)(Aptr + k0);
    float4 a1 = *(const float4*)(Aptr + k0 + 4);
    float4 bv[4];
    #pragma unroll
    for (int i = 0; i < 4; ++i) {
      int flat = t + i * 64;               // 0..255 (256 float4 = 32x32 tile)
      int r  = flat >> 3;                  // K row 0..31
      int cg = (flat & 7) * 4;             // col 0,4,...,28
      bv[i] = *(const float4*)(W + (size_t)(k0 + r) * Nc + n0 + cg);
    }

    __syncthreads();                       // previous iteration done reading LDS

    // ---- stash converted tiles ----
    _Float16* sa = sA + ra * 32 + ca;
    sa[0] = (_Float16)a0.x; sa[1] = (_Float16)a0.y; sa[2] = (_Float16)a0.z; sa[3] = (_Float16)a0.w;
    sa[4] = (_Float16)a1.x; sa[5] = (_Float16)a1.y; sa[6] = (_Float16)a1.z; sa[7] = (_Float16)a1.w;
    #pragma unroll
    for (int i = 0; i < 4; ++i) {
      int flat = t + i * 64;
      int r  = flat >> 3;
      int cg = (flat & 7) * 4;
      // fragment order: elem e = r & 15; lane L = (r>>4)*16 + (n&15); sub-tile w = n>>4
      const int e  = r & 15;
      const float f[4] = {bv[i].x, bv[i].y, bv[i].z, bv[i].w};
      #pragma unroll
      for (int j = 0; j < 4; ++j) {
        int n = cg + j;
        sB[(((n >> 4) * 32 + ((r >> 4) * 16 + (n & 15))) << 4) + e] = (_Float16)f[j];
      }
    }

    __syncthreads();

    // ---- fragments: contiguous LDS reads ----
    // A: elem e -> K = half*16 + e  (row M = l16)
    v8h fa0 = *(const v8h*)(sA + l16 * 32 + half * 16);
    v8h fa1 = *(const v8h*)(sA + l16 * 32 + half * 16 + 8);
    v16h a = __builtin_shufflevector(fa0, fa1, 0, 1, 2, 3, 4, 5, 6, 7,
                                               8, 9, 10, 11, 12, 13, 14, 15);
    v16h b = *(const v16h*)(sB + ((wave * 32 + lane) << 4));

    acc = __builtin_amdgcn_wmma_f32_16x16x32_f16(false, a, false, b, (short)0, acc, false, false);
  }

  // C fragment: VGPR r holds row m0 + r + 8*half, col n0 + wave*16 + l16.
  // M % 16 == 0 -> unguarded straight-line stores; row pointer walks by Nc.
  const int n = n0 + wave * 16 + l16;
  const float bn = bias[n];
  float* cp = C + (size_t)(m0 + 8 * half) * Nc + n;
  #pragma unroll
  for (int r = 0; r < 8; ++r) {
    *cp = acc[r] + bn;
    cp += Nc;
  }
}

// ---------- init out = broadcast bias ----------
__global__ void init_bias_kernel(float* __restrict__ out, const float* __restrict__ bias,
                                 long long total, int HC) {
  long long i = (long long)blockIdx.x * blockDim.x + threadIdx.x;
  if (i < total) out[i] = bias[(int)(i % HC)];
}

// ---------- edge kernels ----------
// logits[e,h] = att[h,:] . leaky_relu(xl[s,h,:] + xr[d,h,:] + ea[e]*We[h,:]); segment max via u32 atomicMax
__global__ void edge_logits_kernel(const int* __restrict__ src, const int* __restrict__ dst,
                                   const float* __restrict__ ea, const float* __restrict__ loopa,
                                   const float* __restrict__ xl, const float* __restrict__ xr,
                                   const float* __restrict__ We, const float* __restrict__ att,
                                   float* __restrict__ logits, unsigned* __restrict__ lmax,
                                   int E, int N, int H, int Cc) {
  long long idx = (long long)blockIdx.x * blockDim.x + threadIdx.x;
  long long total = (long long)(E + N) * H;
  if (idx >= total) return;
  int e = (int)(idx / H), h = (int)(idx % H);
  int s, d; float a;
  if (e < E) { s = src[e]; d = dst[e]; a = ea[e]; }
  else       { s = d = e - E; a = loopa[s]; }
  const int HC = H * Cc;
  const float* xls = xl + (size_t)s * HC + h * Cc;
  const float* xrd = xr + (size_t)d * HC + h * Cc;
  const float* weh = We + h * Cc;
  const float* ath = att + h * Cc;
  float acc = 0.0f;
  for (int c = 0; c < Cc; ++c) {
    float v = xls[c] + xrd[c] + a * weh[c];
    v = (v > 0.0f) ? v : 0.2f * v;          // leaky_relu(0.2)
    acc += ath[c] * v;
  }
  logits[idx] = acc;
  atomicMax(lmax + (size_t)d * H + h, f32key(acc));
}

// p = exp(logit - lmax[d]); den[d] += p  (p overwrites logits)
__global__ void edge_exp_kernel(const int* __restrict__ dst, float* __restrict__ logits,
                                const unsigned* __restrict__ lmax, float* __restrict__ den,
                                int E, int N, int H) {
  long long idx = (long long)blockIdx.x * blockDim.x + threadIdx.x;
  long long total = (long long)(E + N) * H;
  if (idx >= total) return;
  int e = (int)(idx / H), h = (int)(idx % H);
  int d = (e < E) ? dst[e] : (e - E);
  float lm = keyf32(lmax[(size_t)d * H + h]);
  float p = __expf(logits[idx] - lm);
  logits[idx] = p;
  atomicAdd(den + (size_t)d * H + h, p);
}

// out[d,i] += (p[e,h]/den[d,h]) * xl[s,i]
__global__ void edge_scatter_kernel(const int* __restrict__ src, const int* __restrict__ dst,
                                    const float* __restrict__ p, const float* __restrict__ den,
                                    const float* __restrict__ xl, float* __restrict__ out,
                                    int E, int N, int H, int Cc) {
  const int HC = H * Cc;
  long long idx = (long long)blockIdx.x * blockDim.x + threadIdx.x;
  long long total = (long long)(E + N) * HC;
  if (idx >= total) return;
  int e = (int)(idx / HC), i = (int)(idx % HC);
  int h = i / Cc;
  int s, d;
  if (e < E) { s = src[e]; d = dst[e]; } else { s = d = e - E; }
  float alpha = p[(size_t)e * H + h] / den[(size_t)d * H + h];
  atomicAdd(out + (size_t)d * HC + i, alpha * xl[(size_t)s * HC + i]);
}

__global__ void relu_kernel(float* __restrict__ x, long long n) {
  long long i = (long long)blockIdx.x * blockDim.x + threadIdx.x;
  if (i < n) x[i] = fmaxf(x[i], 0.0f);
}

// ---------- global mean pool ----------
__global__ void pool_kernel(const float* __restrict__ hp, const int* __restrict__ batch,
                            float* __restrict__ gs, float* __restrict__ gc, int N) {
  long long idx = (long long)blockIdx.x * blockDim.x + threadIdx.x;
  if (idx >= (long long)N * 32) return;
  int n = (int)(idx >> 5), c = (int)(idx & 31);
  int g = batch[n];
  atomicAdd(gs + (size_t)g * 32 + c, hp[idx]);
  if (c == 0) atomicAdd(gc + g, 1.0f);
}
__global__ void pool_final_kernel(const float* __restrict__ gs, const float* __restrict__ gc,
                                  float* __restrict__ out, int total) {
  int i = blockIdx.x * blockDim.x + threadIdx.x;
  if (i < total) out[i] = gs[i] / fmaxf(gc[i >> 5], 1.0f);
}

static inline int cdiv_ll(long long a, long long b) { return (int)((a + b - 1) / b); }

extern "C" void kernel_launch(void* const* d_in, const int* in_sizes, int n_in,
                              void* d_out, int out_size, void* d_ws, size_t ws_size,
                              hipStream_t stream) {
  const float* x    = (const float*)d_in[0];
  const int*   ei   = (const int*)d_in[1];
  const float* ea   = (const float*)d_in[2];
  const int*   batch= (const int*)d_in[3];
  const float* Wl1  = (const float*)d_in[4];
  const float* bl1  = (const float*)d_in[5];
  const float* Wr1  = (const float*)d_in[6];
  const float* br1  = (const float*)d_in[7];
  const float* We1  = (const float*)d_in[8];
  const float* att1 = (const float*)d_in[9];
  const float* b1   = (const float*)d_in[10];
  const float* Wl2  = (const float*)d_in[11];
  const float* bl2  = (const float*)d_in[12];
  const float* Wr2  = (const float*)d_in[13];
  const float* br2  = (const float*)d_in[14];
  const float* We2  = (const float*)d_in[15];
  const float* att2 = (const float*)d_in[16];
  const float* b2   = (const float*)d_in[17];
  const float* Wp   = (const float*)d_in[18];
  const float* bp   = (const float*)d_in[19];

  const int N  = in_sizes[3];            // batch has N elements
  const int E  = in_sizes[2];            // edge_attr is [E,1]
  const int F  = in_sizes[0] / N;        // 128
  const int G  = out_size / 32;          // 64
  const int H1 = 4, C1 = 64, HC1 = 256;
  const int H2 = 2, C2 = 32, HC2 = 64;
  const int P  = 32;

  const int* srcp = ei;        // edge_index row 0
  const int* dstp = ei + E;    // edge_index row 1

  // ---- workspace layout (~166 MB) ----
  char* ws = (char*)d_ws;
  float* xl1    = (float*)(ws);                                   // N*HC1
  float* xr1    = (float*)(ws + (size_t)N * HC1 * 4);             // N*HC1
  float* h      = (float*)(ws + (size_t)2 * N * HC1 * 4);         // N*HC1 (conv1 out -> relu)
  float* logits = (float*)(ws + (size_t)3 * N * HC1 * 4);         // (E+N)*H1
  size_t off = (size_t)3 * N * HC1 * 4 + (size_t)(E + N) * H1 * 4;
  const size_t off_small = off;
  float*    sum_a = (float*)(ws + off);    off += (size_t)N * 4;
  float*    cnt_a = (float*)(ws + off);    off += (size_t)N * 4;
  float*    loopa = (float*)(ws + off);    off += (size_t)N * 4;
  unsigned* lmax  = (unsigned*)(ws + off); off += (size_t)N * H1 * 4;
  float*    den   = (float*)(ws + off);    off += (size_t)N * H1 * 4;
  float*    gs    = (float*)(ws + off);    off += (size_t)G * P * 4;
  float*    gc    = (float*)(ws + off);    off += (size_t)G * 4;
  // layer-2 buffers overlay the (dead after layer-1 scatter) xl1 region
  float* xl2  = xl1;
  float* xr2  = (float*)((char*)xl1 + (size_t)N * HC2 * 4);
  float* out2 = (float*)((char*)xl1 + (size_t)2 * N * HC2 * 4);
  float* hp   = (float*)((char*)xl1 + (size_t)3 * N * HC2 * 4);   // N*P

  const int TB = 256;
  // 0) zero all accumulator/state buffers (sum..gc contiguous)
  {
    long long zw = (long long)(off - off_small) / 4;
    zero_u32_kernel<<<cdiv_ll(zw, TB), TB, 0, stream>>>((unsigned*)sum_a, zw);
  }
  // 1) self-loop edge attr = mean of incoming
  loop_sum_kernel<<<cdiv_ll(E, TB), TB, 0, stream>>>(dstp, ea, sum_a, cnt_a, E);
  loop_div_kernel<<<cdiv_ll(N, TB), TB, 0, stream>>>(sum_a, cnt_a, loopa, N);
  // 2) layer-1 linear transforms (WMMA), 16x32 tiles
  gemm_wmma_kernel<<<dim3(HC1 / 32, cdiv_ll(N, 16)), 64, 0, stream>>>(x, Wl1, bl1, xl1, N, F, HC1);
  gemm_wmma_kernel<<<dim3(HC1 / 32, cdiv_ll(N, 16)), 64, 0, stream>>>(x, Wr1, br1, xr1, N, F, HC1);
  // 3) layer-1 attention + aggregation
  init_bias_kernel<<<cdiv_ll((long long)N * HC1, TB), TB, 0, stream>>>(h, b1, (long long)N * HC1, HC1);
  {
    long long tEH  = (long long)(E + N) * H1;
    long long tEHC = (long long)(E + N) * HC1;
    edge_logits_kernel<<<cdiv_ll(tEH, TB), TB, 0, stream>>>(srcp, dstp, ea, loopa, xl1, xr1, We1, att1,
                                                            logits, lmax, E, N, H1, C1);
    edge_exp_kernel<<<cdiv_ll(tEH, TB), TB, 0, stream>>>(dstp, logits, lmax, den, E, N, H1);
    edge_scatter_kernel<<<cdiv_ll(tEHC, TB), TB, 0, stream>>>(srcp, dstp, logits, den, xl1, h, E, N, H1, C1);
  }
  relu_kernel<<<cdiv_ll((long long)N * HC1, TB), TB, 0, stream>>>(h, (long long)N * HC1);
  // 4) reset softmax state for layer 2 (lmax+den contiguous = 8N words)
  zero_u32_kernel<<<cdiv_ll(8LL * N, TB), TB, 0, stream>>>(lmax, 8LL * N);
  // 5) layer-2 linear transforms (WMMA)
  gemm_wmma_kernel<<<dim3(HC2 / 32, cdiv_ll(N, 16)), 64, 0, stream>>>(h, Wl2, bl2, xl2, N, HC1, HC2);
  gemm_wmma_kernel<<<dim3(HC2 / 32, cdiv_ll(N, 16)), 64, 0, stream>>>(h, Wr2, br2, xr2, N, HC1, HC2);
  // 6) layer-2 attention + aggregation
  init_bias_kernel<<<cdiv_ll((long long)N * HC2, TB), TB, 0, stream>>>(out2, b2, (long long)N * HC2, HC2);
  {
    long long tEH  = (long long)(E + N) * H2;
    long long tEHC = (long long)(E + N) * HC2;
    edge_logits_kernel<<<cdiv_ll(tEH, TB), TB, 0, stream>>>(srcp, dstp, ea, loopa, xl2, xr2, We2, att2,
                                                            logits, lmax, E, N, H2, C2);
    edge_exp_kernel<<<cdiv_ll(tEH, TB), TB, 0, stream>>>(dstp, logits, lmax, den, E, N, H2);
    edge_scatter_kernel<<<cdiv_ll(tEHC, TB), TB, 0, stream>>>(srcp, dstp, logits, den, xl2, out2, E, N, H2, C2);
  }
  // 7) projection (WMMA) + global mean pool
  gemm_wmma_kernel<<<dim3(P / 32, cdiv_ll(N, 16)), 64, 0, stream>>>(out2, Wp, bp, hp, N, HC2, P);
  pool_kernel<<<cdiv_ll((long long)N * P, TB), TB, 0, stream>>>(hp, batch, gs, gc, N);
  pool_final_kernel<<<cdiv_ll(G * P, TB), TB, 0, stream>>>(gs, gc, (float*)d_out, G * P);
}
